// Decoder_661424964037
// MI455X (gfx1250) — matrix operand order repaired and checked by reading, not compiled, not run on hardware
//
#include <hip/hip_runtime.h>

// -------------------------------------------------------------------------
// CDNA5 (gfx1250) decoder. All convs (incl. transposed conv via zero-
// dilation) are implicit GEMM on v_wmma_f32_16x16x32_bf16 (wave32):
//   M = Cout, N = B*OH*OW, K = Cin*KH*KW.
// Block = 4 waves = 4 M-tiles (64 rows) x one 32-pixel N-tile. Each wave
// holds two 16x16 accumulators (A-fragment reuse across 2 WMMAs). B-chunks
// are staged in LDS as bf16 [n][k]; weights pre-packed to bf16
// [Co][kh][kw][Ci] so A-fragments are contiguous 16B loads. Next-chunk
// gather addresses are prefetched (global_prefetch_b8).
// -------------------------------------------------------------------------

#define ACT_NONE  0
#define ACT_RELU  1
#define ACT_PRELU 2

typedef __attribute__((ext_vector_type(16))) __bf16 v16bf;
typedef __attribute__((ext_vector_type(8)))  float  v8f;

template<int KH, int KW>
__global__ __launch_bounds__(128) void conv_wmma_t(
    const float* __restrict__ in, const __bf16* __restrict__ wpk,
    const float* __restrict__ bias, const float* __restrict__ alpha,
    const float* __restrict__ resid, float* __restrict__ out,
    int B, int Ci, int H, int W, int Co, int pad, int OH, int OW, int act)
{
  constexpr int KHW = KH * KW;
  __shared__ __bf16 lds_b[32][32];            // B chunk, [n][k]: lane reads two 16B runs

  const int lane  = threadIdx.x;              // 0..31 (wave32)
  const int wave  = threadIdx.y;              // 0..3 -> M-tile within block
  const int tid   = lane + (wave << 5);       // 0..127
  const int khalf = lane >> 4;
  const int m0    = blockIdx.x * 64 + wave * 16;
  const int n0    = blockIdx.y * 32;          // 32-pixel N-tile per block
  const int Np    = B * OH * OW;
  const int HW    = H * W;

  // ---- fill role: thread fills column n_f, k = k0..k0+7 of the LDS chunk -
  const int  n_f  = tid >> 2;                 // 0..31
  const int  k0   = (tid & 3) << 3;           // 0,8,16,24
  const int  ncf  = n0 + n_f;
  const bool f_in = ncf < Np;
  int bF = 0, oyF = 0, oxF = 0;
  if (f_in) { bF = ncf / (OH * OW); int r = ncf - bF * (OH * OW); oyF = r / OW; oxF = r - oyF * OW; }
  const float* fbase = in + (size_t)bF * Ci * HW;

  // ---- A row: lane owns M = m0 + (lane&15); OOB rows read row 0 (results
  //      are masked at store time, so garbage is never written) -----------
  const int mrow = m0 + (lane & 15);
  const __bf16* wrow = wpk + (size_t)(mrow < Co ? mrow : 0) * KHW * Ci;

  v8f acc0 = {}, acc1 = {};

#pragma unroll
  for (int kh = 0; kh < KH; ++kh) {
#pragma unroll
    for (int kw = 0; kw < KW; ++kw) {
      const int  iy   = oyF - pad + kh;
      const int  ix   = oxF - pad + kw;
      const bool f_ok = f_in && iy >= 0 && iy < H && ix >= 0 && ix < W;
      const float*  src = fbase + (size_t)iy * W + ix;
      const __bf16* wk  = wrow + (kh * KW + kw) * Ci;

      for (int cc = 0; cc < Ci; cc += 32) {      // Ci is a multiple of 32
        __syncthreads();                          // protect LDS reuse (WAR)
#pragma unroll
        for (int j = 0; j < 8; ++j) {
          float v = f_ok ? src[(size_t)(cc + k0 + j) * HW] : 0.f;
          lds_b[n_f][k0 + j] = (__bf16)v;
        }
        if (f_ok && cc + 32 < Ci)                 // hide next chunk's gather
          __builtin_prefetch(src + (size_t)(cc + 32 + k0) * HW, 0, 1);
        __syncthreads();

        // A fragment (16-bit A 16x32, ISA 7.12.2):
        // lanes 0-15: K {0..7,16..23}; lanes 16-31: K {8..15,24..31} -> two
        // contiguous 8-elem (16B) bf16 runs of the packed weight row.
        v16bf a;
#pragma unroll
        for (int i = 0; i < 8; ++i) a[i]     = wk[cc + khalf * 8 + i];
#pragma unroll
        for (int i = 0; i < 8; ++i) a[8 + i] = wk[cc + 16 + khalf * 8 + i];

        // B fragments: lanes 0-15 K {0..7,16..23}, lanes 16-31 K {8..15,24..31}, N = lane&15
        v16bf bv0, bv1;
#pragma unroll
        for (int i = 0; i < 16; ++i) bv0[i] = lds_b[lane & 15][khalf * 8 + (i & 7) + ((i >> 3) << 4)];
#pragma unroll
        for (int i = 0; i < 16; ++i) bv1[i] = lds_b[16 + (lane & 15)][khalf * 8 + (i & 7) + ((i >> 3) << 4)];

        acc0 = __builtin_amdgcn_wmma_f32_16x16x32_bf16(
            false, a, false, bv0, (short)0, acc0, false, false);
        acc1 = __builtin_amdgcn_wmma_f32_16x16x32_bf16(
            false, a, false, bv1, (short)0, acc1, false, false);
      }
    }
  }

  // ---- epilogue: C/D layout VGPR r -> M = m0 + r + khalf*8, N = lane&15 --
  const float al = (act == ACT_PRELU && alpha) ? alpha[0] : 0.f;
#pragma unroll
  for (int t = 0; t < 2; ++t) {
    const v8f& acc = t ? acc1 : acc0;
    const int nc = n0 + t * 16 + (lane & 15);
    if (nc < Np) {
      int b = nc / (OH * OW); int rr = nc - b * (OH * OW);
      int oy = rr / OW;       int ox = rr - oy * OW;
#pragma unroll
      for (int r = 0; r < 8; ++r) {
        int m = m0 + r + khalf * 8;
        if (m < Co) {
          float v = acc[r];
          if (bias) v += bias[m];
          if (act == ACT_RELU)       v = v > 0.f ? v : 0.f;
          else if (act == ACT_PRELU) v = v >= 0.f ? v : al * v;
          size_t o = (((size_t)b * Co + m) * OH + oy) * OW + ox;
          if (resid) v += resid[o];
          out[o] = v;
        }
      }
    }
  }
}

// -------- weight pack: OIHW f32 -> [Co][kh][kw][Ci] bf16 ------------------
__global__ void pack_w_kernel(const float* __restrict__ w, __bf16* __restrict__ wp,
                              int Co, int Ci, int KHW)
{
  size_t idx = (size_t)blockIdx.x * blockDim.x + threadIdx.x;
  size_t total = (size_t)Co * Ci * KHW;
  if (idx >= total) return;
  int c = idx % Ci; size_t t = idx / Ci;
  int r = t % KHW;  int m = t / KHW;           // idx = (m*KHW + r)*Ci + c
  wp[idx] = (__bf16)w[((size_t)m * Ci + c) * KHW + r];
}

// -------- ConvTranspose weight [Ci][Co][3][3] -> flipped packed bf16 ------
__global__ void pack_tconv_kernel(const float* __restrict__ w, __bf16* __restrict__ wp,
                                  int C)  // Ci == Co == C
{
  size_t idx = (size_t)blockIdx.x * blockDim.x + threadIdx.x;
  size_t total = (size_t)C * C * 9;
  if (idx >= total) return;
  int i = idx % C; size_t t = idx / C;
  int kw = t % 3; t /= 3;
  int kh = t % 3; int o = t / 3;               // idx = ((o*3+kh)*3+kw)*C + i
  wp[idx] = (__bf16)w[(((size_t)i * C + o) * 3 + (2 - kh)) * 3 + (2 - kw)];
}

// ---------------- filter-adaptive conv (edge pad, 5x5, leaky 0.2) ---------
__global__ void fac_conv_kernel(const float* __restrict__ x,
                                const float* __restrict__ kern,
                                float* __restrict__ out,
                                int B, int C, int H, int W)
{
  size_t idx = (size_t)blockIdx.x * blockDim.x + threadIdx.x;
  size_t total = (size_t)B * C * H * W;
  if (idx >= total) return;
  int xw = idx % W; size_t t = idx / W;
  int yy = t % H;  t /= H;
  int c  = t % C;  int b = t / C;

  const float* xin = x + ((size_t)b * C + c) * H * W;
  const float* kp  = kern + ((((size_t)b * C + c) * 25) * H + yy) * W + xw;
  const size_t kstride = (size_t)H * W;

  float s = 0.f; int tap = 0;
#pragma unroll
  for (int i = 0; i < 5; ++i) {
    int iy = yy + i - 2; iy = iy < 0 ? 0 : (iy >= H ? H - 1 : iy);
#pragma unroll
    for (int j = 0; j < 5; ++j, ++tap) {
      int ix = xw + j - 2; ix = ix < 0 ? 0 : (ix >= W ? W - 1 : ix);
      s += xin[(size_t)iy * W + ix] * kp[tap * kstride];
    }
  }
  out[idx] = s >= 0.f ? s : 0.2f * s;
}

// ---------------- bilinear x2 upsample (align_corners=False) --------------
__global__ void upsample2_kernel(const float* __restrict__ in,
                                 float* __restrict__ out,
                                 int B, int C, int H, int W)
{
  int OH = 2 * H, OW = 2 * W;
  size_t idx = (size_t)blockIdx.x * blockDim.x + threadIdx.x;
  size_t total = (size_t)B * C * OH * OW;
  if (idx >= total) return;
  int ox = idx % OW; size_t t = idx / OW;
  int oy = t % OH;  t /= OH;
  int c  = t % C;   int b = t / C;

  float fy = (oy + 0.5f) * 0.5f - 0.5f;
  float fx = (ox + 0.5f) * 0.5f - 0.5f;
  int y0 = (int)floorf(fy), x0 = (int)floorf(fx);
  float wy = fy - y0, wx = fx - x0;
  int y0c = y0 < 0 ? 0 : (y0 >= H ? H - 1 : y0);
  int y1c = y0 + 1 < 0 ? 0 : (y0 + 1 >= H ? H - 1 : y0 + 1);
  int x0c = x0 < 0 ? 0 : (x0 >= W ? W - 1 : x0);
  int x1c = x0 + 1 < 0 ? 0 : (x0 + 1 >= W ? W - 1 : x0 + 1);

  const float* p = in + ((size_t)b * C + c) * H * W;
  float v00 = p[(size_t)y0c * W + x0c], v01 = p[(size_t)y0c * W + x1c];
  float v10 = p[(size_t)y1c * W + x0c], v11 = p[(size_t)y1c * W + x1c];
  float v0 = v00 + (v01 - v00) * wx;
  float v1 = v10 + (v11 - v10) * wx;
  out[idx] = v0 + (v1 - v0) * wy;
}

// -------- lhs-dilation(2) + pad(1,2) for ConvTranspose(3, s2, p1, op1) ----
__global__ void dilate_pad_kernel(const float* __restrict__ in,
                                  float* __restrict__ out,
                                  int B, int C, int H, int W)
{
  int OH = 2 * H + 2, OW = 2 * W + 2;
  size_t idx = (size_t)blockIdx.x * blockDim.x + threadIdx.x;
  size_t total = (size_t)B * C * OH * OW;
  if (idx >= total) return;
  int ox = idx % OW; size_t t = idx / OW;
  int oy = t % OH;  t /= OH;
  int c  = t % C;   int b = t / C;

  float v = 0.f;
  int ry = oy - 1, rx = ox - 1;
  if (ry >= 0 && rx >= 0 && !(ry & 1) && !(rx & 1)) {
    int iy = ry >> 1, ix = rx >> 1;
    if (iy < H && ix < W)
      v = in[(((size_t)b * C + c) * H + iy) * W + ix];
  }
  out[idx] = v;
}

// -------------------------- host-side helpers -----------------------------
static void run_conv(hipStream_t s, const float* in, const __bf16* wp,
                     const float* bias, const float* alpha, const float* resid,
                     float* out, int B, int Ci, int H, int W, int Co,
                     int KH, int KW, int pad, int act)
{
  int OH = H + 2 * pad - KH + 1;
  int OW = W + 2 * pad - KW + 1;
  int Np = B * OH * OW;
  dim3 grid((Co + 63) / 64, (Np + 31) / 32);
  dim3 block(32, 4);
  if (KH == 1)
    conv_wmma_t<1, 1><<<grid, block, 0, s>>>(in, wp, bias, alpha, resid, out,
                                             B, Ci, H, W, Co, pad, OH, OW, act);
  else
    conv_wmma_t<3, 3><<<grid, block, 0, s>>>(in, wp, bias, alpha, resid, out,
                                             B, Ci, H, W, Co, pad, OH, OW, act);
}

static inline int ew_blocks(size_t total) { return (int)((total + 255) / 256); }

extern "C" void kernel_launch(void* const* d_in, const int* in_sizes, int n_in,
                              void* d_out, int out_size, void* d_ws, size_t ws_size,
                              hipStream_t stream)
{
  (void)in_sizes; (void)out_size; (void)ws_size;
  const int B = 2;

  const float* x  = (const float*)d_in[0];   // (2,128,32,32)
  const float* x8 = (const float*)d_in[1];   // (2,128,32,32)
  const float* x4 = (const float*)d_in[2];   // (2,128,64,64)
  const float* x2 = (const float*)d_in[3];   // (2,64,128,128)

  const float* P[64];
  int np = n_in - 4; if (np > 64) np = 64;
  for (int i = 0; i < np; ++i) P[i] = (const float*)d_in[4 + i];
  // P layout (setup_inputs insertion order):
  //  0..7  fac3 w1,b1,w2,b2,w3,b3,w4,b4     8..15 fac2   16..23 fac1
  // 24..35 db3 t1,tw,t2,b1,c1a,c1b,c2a,c2b,tp1,tp2,p1,p2
  // 36..47 db2                              48..59 db1
  // 60 final_w, 61 final_b
  const int FAC3 = 0, FAC2 = 8, FAC1 = 16, DB3 = 24, DB2 = 36, DB1 = 48;

  // -------- bump allocator over d_ws --------------------------------------
  char* base = (char*)d_ws;
  size_t off = 0;
  auto allocB = [&](size_t nbytes) -> void* {
    void* p = base + off;
    off = (off + nbytes + 255) & ~((size_t)255);
    return p;
  };
  auto alloc = [&](size_t nflt) -> float* { return (float*)allocB(nflt * 4); };

  float* carryA = alloc(4194304);   // max carried tensor: (2,32,256,256)
  float* carryB = alloc(4194304);
  const size_t stage_base = off;

  // pack + conv helper (pack buffer lives in current stage scope)
  auto pconv = [&](const float* in, const float* w, const float* bias,
                   const float* alpha, const float* resid, float* out,
                   int Ci, int H, int W, int Co, int KH, int KW, int pad, int act) {
    size_t nw = (size_t)Co * Ci * KH * KW;
    __bf16* wp = (__bf16*)allocB(nw * 2);
    pack_w_kernel<<<ew_blocks(nw), 256, 0, stream>>>(w, wp, Co, Ci, KH * KW);
    run_conv(stream, in, wp, bias, alpha, resid, out, B, Ci, H, W, Co, KH, KW, pad, act);
  };

  // ======================= stage helpers ==================================
  auto fac_stage = [&](const float* src, const float* ximg, int C, int H, int W,
                       int pbase, float* dst) {
    off = stage_base;
    size_t npx = (size_t)B * C * H * W;
    float* a1 = alloc(npx);
    float* a2 = alloc(npx);
    float* kern = alloc((size_t)B * C * 25 * H * W);
    pconv(src, P[pbase + 0], P[pbase + 1], nullptr, nullptr, a1, C, H, W, C, 3, 3, 1, ACT_RELU);
    pconv(a1,  P[pbase + 2], P[pbase + 3], nullptr, nullptr, a2, C, H, W, C, 3, 3, 1, ACT_RELU);
    pconv(a2,  P[pbase + 4], P[pbase + 5], nullptr, nullptr, a1, C, H, W, C, 3, 3, 1, ACT_RELU);
    pconv(a1,  P[pbase + 6], P[pbase + 7], nullptr, nullptr, kern, C, H, W, C * 25, 1, 1, 0, ACT_NONE);
    fac_conv_kernel<<<ew_blocks(npx), 256, 0, stream>>>(ximg, kern, dst, B, C, H, W);
  };

  auto decoder_block = [&](const float* src, int Ci, int Co, int H, int W,
                           int pbase, float* dst) {
    off = stage_base;
    int OH = 2 * H, OW = 2 * W;
    size_t n_in_px  = (size_t)B * Ci * H * W;
    size_t n_mid_px = (size_t)B * Ci * OH * OW;
    size_t n_out_px = (size_t)B * Co * OH * OW;
    size_t n_dil    = (size_t)B * Ci * (OH + 2) * (OW + 2);

    const float* t1 = P[pbase + 0], *tw = P[pbase + 1], *t2 = P[pbase + 2];
    const float* b1 = P[pbase + 3];
    const float* c1a = P[pbase + 4], *c1b = P[pbase + 5];
    const float* c2a = P[pbase + 6], *c2b = P[pbase + 7];
    const float* tp1 = P[pbase + 8], *tp2 = P[pbase + 9];
    const float* p1  = P[pbase + 10], *p2 = P[pbase + 11];

    float* t_a = alloc(n_in_px);
    float* dil = alloc(n_dil);
    float* t_b = alloc(n_mid_px);
    float* t_c = alloc(n_out_px);
    float* up  = alloc(n_mid_px);
    float* xb  = alloc(n_out_px);
    float* r   = alloc(n_out_px);
    float* xb2 = alloc(n_out_px);

    // ResidualUpsample main path
    pconv(src, t1, nullptr, tp1, nullptr, t_a, Ci, H, W, Ci, 1, 1, 0, ACT_PRELU);
    dilate_pad_kernel<<<ew_blocks(n_dil), 256, 0, stream>>>(t_a, dil, B, Ci, H, W);
    {
      size_t ntw = (size_t)Ci * Ci * 9;
      __bf16* wpt = (__bf16*)allocB(ntw * 2);
      pack_tconv_kernel<<<ew_blocks(ntw), 256, 0, stream>>>(tw, wpt, Ci);
      run_conv(stream, dil, wpt, nullptr, tp2, nullptr, t_b,
               B, Ci, OH + 2, OW + 2, Ci, 3, 3, 0, ACT_PRELU);
    }
    pconv(t_b, t2, nullptr, nullptr, nullptr, t_c, Ci, OH, OW, Co, 1, 1, 0, ACT_NONE);
    // skip path + add
    upsample2_kernel<<<ew_blocks(n_mid_px), 256, 0, stream>>>(src, up, B, Ci, H, W);
    pconv(up, b1, nullptr, nullptr, t_c, xb, Ci, OH, OW, Co, 1, 1, 0, ACT_NONE);
    // ResidualBlock2 #1
    pconv(xb, c1a, nullptr, p1, nullptr, r,   Co, OH, OW, Co, 3, 3, 1, ACT_PRELU);
    pconv(r,  c1b, nullptr, nullptr, xb, xb2, Co, OH, OW, Co, 3, 3, 1, ACT_NONE);
    // ResidualBlock2 #2
    pconv(xb2, c2a, nullptr, p2, nullptr, r,  Co, OH, OW, Co, 3, 3, 1, ACT_PRELU);
    pconv(r,  c2b, nullptr, nullptr, xb2, dst, Co, OH, OW, Co, 3, 3, 1, ACT_NONE);
  };

  // =============================== forward ================================
  fac_stage(x8, x, 128, 32, 32, FAC3, carryA);
  decoder_block(carryA, 128, 128, 32, 32, DB3, carryB);

  fac_stage(x4, carryB, 128, 64, 64, FAC2, carryA);
  decoder_block(carryA, 128, 64, 64, 64, DB2, carryB);

  fac_stage(x2, carryB, 64, 128, 128, FAC1, carryA);
  decoder_block(carryA, 64, 32, 128, 128, DB1, carryB);

  // final 3x3 conv 32 -> 3 @ 256x256
  off = stage_base;
  pconv(carryB, P[60], P[61], nullptr, nullptr, (float*)d_out,
        32, 256, 256, 3, 3, 3, 1, ACT_NONE);
}
